// FullAttentionMHA_56530359550728
// MI455X (gfx1250) — matrix operand order repaired and checked by reading, not compiled, not run on hardware
//
#include <hip/hip_runtime.h>
#include <hip/hip_bf16.h>
#include <math.h>

#define HEADS  8
#define DH     64
#define WIN    256
#define NB     2
#define SEQ    2048
#define DMODEL 512
#define INNER  512            // HEADS*DH
#define NQKV   1536
#define BS     (NB*SEQ)       // 4096
#define NT     18             // KV tiles per row-tile band (covers window 256 + diag tile)
#define LDP    296            // P-band LDS row stride in elements (296*2=592, 16B aligned)
#define VBW    288            // max V-band width (columns)
#define GM     2              // GEMM wave tile: rows of 16x16 tiles
#define GN     4              // GEMM wave tile: cols of 16x16 tiles

#if __has_builtin(__builtin_amdgcn_tensor_load_to_lds)
#define USE_TDM 1
#else
#define USE_TDM 0
#endif

typedef __attribute__((ext_vector_type(16))) __bf16 v16bf;
typedef __attribute__((ext_vector_type(8)))  __bf16 v8bf;
typedef __attribute__((ext_vector_type(8)))  float  v8f;
typedef __attribute__((ext_vector_type(4)))  unsigned int v4u;
typedef __attribute__((ext_vector_type(8)))  int v8i;
typedef __attribute__((ext_vector_type(4)))  int v4i;

// ---- WMMA fragment loaders (contiguous k order shared by A and B, wave32) ----------
// A (16x32, MxK): lane m = lane&15, half = lane>>4.
//   elem e <-> K = half*16 + e
static __device__ inline v16bf frag_rowK(const __bf16* base, int ld, int lane) {
  const int m = lane & 15, h = lane >> 4;
  const __bf16* p = base + (size_t)m * ld + h * 16;
  v8bf lo = *(const v8bf*)(p);
  v8bf hi = *(const v8bf*)(p + 8);
  v16bf r;
#pragma unroll
  for (int e = 0; e < 8; ++e) { r[e] = lo[e]; r[8 + e] = hi[e]; }
  return r;
}
// B (32x16, KxN) from n-major storage Bt[n][k]: elem e <-> K = half*16 + e
static __device__ inline v16bf frag_colN(const __bf16* base, int ld, int lane) {
  const int n = lane & 15, h = lane >> 4;
  const __bf16* p = base + (size_t)n * ld + h * 16;
  v8bf lo = *(const v8bf*)(p);
  v8bf hi = *(const v8bf*)(p + 8);
  v16bf r;
#pragma unroll
  for (int e = 0; e < 8; ++e) { r[e] = lo[e]; r[8 + e] = hi[e]; }
  return r;
}

static __device__ inline v8f wmma_bf16(v16bf a, v16bf b, v8f c) {
  return __builtin_amdgcn_wmma_f32_16x16x32_bf16(false, a, false, b, (short)0, c,
                                                 false, false);
}

// ---- Stage 0: format conversion -------------------------------------------
__global__ void f32_to_bf16(const float* __restrict__ in, __bf16* __restrict__ out, int n) {
  int i = blockIdx.x * blockDim.x + threadIdx.x;
  int stride = gridDim.x * blockDim.x;
  for (; i < n; i += stride) out[i] = (__bf16)in[i];
}

// in: K x N row-major fp32 -> out: N x K row-major bf16
__global__ void transpose_bf16(const float* __restrict__ in, __bf16* __restrict__ out,
                               int K, int N) {
  int idx = blockIdx.x * blockDim.x + threadIdx.x;
  int total = K * N;
  int stride = gridDim.x * blockDim.x;
  for (; idx < total; idx += stride) {
    int k = idx / N, n = idx % N;
    out[(size_t)n * K + k] = (__bf16)in[idx];
  }
}

__global__ void zero_f4(float4* __restrict__ p, size_t n4) {
  size_t i = (size_t)blockIdx.x * blockDim.x + threadIdx.x;
  size_t stride = (size_t)gridDim.x * blockDim.x;
  float4 z = make_float4(0.f, 0.f, 0.f, 0.f);
  for (; i < n4; i += stride) p[i] = z;
}

// ---- Stage 1: QKV projection, 32x64 per wave (2x4 WMMA tiles) --------------
// q:(B,H,S,Dh) pre-scaled by 1/sqrt(Dh); k:(B,H,S,Dh); v transposed (B,H,Dh,S)
__global__ __launch_bounds__(32)
void gemm_qkv(const __bf16* __restrict__ xb, const __bf16* __restrict__ wT,
              __bf16* __restrict__ qb, __bf16* __restrict__ kb,
              __bf16* __restrict__ vtb) {
  const int lane = threadIdx.x;
  const int m0 = blockIdx.x * (16 * GM);
  const int n0 = blockIdx.y * (16 * GN);
  v8f c[GM][GN] = {};
  for (int k0 = 0; k0 < DMODEL; k0 += 32) {
    __builtin_prefetch(xb + (size_t)m0 * DMODEL + k0 + 64, 0, 1);
    v16bf a[GM], b[GN];
#pragma unroll
    for (int mt = 0; mt < GM; ++mt)
      a[mt] = frag_rowK(xb + (size_t)(m0 + 16 * mt) * DMODEL + k0, DMODEL, lane);
#pragma unroll
    for (int nt = 0; nt < GN; ++nt)
      b[nt] = frag_colN(wT + (size_t)(n0 + 16 * nt) * DMODEL + k0, DMODEL, lane);
#pragma unroll
    for (int mt = 0; mt < GM; ++mt)
#pragma unroll
      for (int nt = 0; nt < GN; ++nt)
        c[mt][nt] = wmma_bf16(a[mt], b[nt], c[mt][nt]);
  }
  // Epilogue: which/h/d0 are *tile-uniform* (each 16-wide N tile lies inside
  // one 64-aligned head segment of one 512-aligned q/k/v segment) -> scalar
  // branch per tile, coalesced stores inside.
  const int half = lane >> 4;
  const int nn = lane & 15;
#pragma unroll
  for (int nt = 0; nt < GN; ++nt) {
    const int nb = n0 + nt * 16;     // uniform
    const int which = nb >> 9;       // uniform: 0=q 1=k 2=v
    const int rem = nb & 511;        // uniform
    const int h = rem >> 6;          // uniform
    const int d = (rem & 63) + nn;   // d-tile base uniform + lane
    if (which == 0) {
#pragma unroll
      for (int mt = 0; mt < GM; ++mt)
#pragma unroll
        for (int v = 0; v < 8; ++v) {
          const int row = m0 + mt * 16 + v + 8 * half;
          const int bb = row >> 11, s = row & (SEQ - 1);
          qb[(((size_t)(bb * HEADS + h)) * SEQ + s) * DH + d] =
              (__bf16)(c[mt][nt][v] * 0.125f);
        }
    } else if (which == 1) {
#pragma unroll
      for (int mt = 0; mt < GM; ++mt)
#pragma unroll
        for (int v = 0; v < 8; ++v) {
          const int row = m0 + mt * 16 + v + 8 * half;
          const int bb = row >> 11, s = row & (SEQ - 1);
          kb[(((size_t)(bb * HEADS + h)) * SEQ + s) * DH + d] = (__bf16)c[mt][nt][v];
        }
    } else {
#pragma unroll
      for (int mt = 0; mt < GM; ++mt)
#pragma unroll
        for (int v = 0; v < 8; ++v) {
          const int row = m0 + mt * 16 + v + 8 * half;
          const int bb = row >> 11, s = row & (SEQ - 1);
          vtb[(((size_t)(bb * HEADS + h)) * DH + d) * SEQ + s] = (__bf16)c[mt][nt][v];
        }
    }
  }
}

// ---- Stage 2: banded attention, one wave per (b,h,16-row tile) -------------
__global__ __launch_bounds__(32)
void attn_kernel(const __bf16* __restrict__ qb, const __bf16* __restrict__ kb,
                 const __bf16* __restrict__ vtb, float* __restrict__ attn,
                 __bf16* __restrict__ ctx) {
  __shared__ __bf16 pband[16 * LDP];
  __shared__ __bf16 vband[64 * VBW];
  const int lane = threadIdx.x;
  int bid = blockIdx.x;
  const int ti = bid & 127; bid >>= 7;
  const int h  = bid & 7;   bid >>= 3;
  const int b  = bid;
  const int i0 = ti * 16;
  const int half = lane >> 4, nn = lane & 15;

  const __bf16* qh  = qb  + ((size_t)(b * HEADS + h)) * SEQ * DH;
  const __bf16* kh  = kb  + ((size_t)(b * HEADS + h)) * SEQ * DH;
  const __bf16* vth = vtb + ((size_t)(b * HEADS + h)) * DH * SEQ;
  float* ah = attn + ((size_t)(b * HEADS + h)) * SEQ * SEQ;

  const int jstart = i0 - (NT - 1) * 16;          // i0 - 272
  const int jlo    = jstart < 0 ? 0 : jstart;
  const int width  = i0 + 16 - jlo;               // <= 288, multiple of 16

#if USE_TDM
  // Tensor Data Mover copy of the V band (64 x width bf16, rows strided by
  // SEQ) into LDS; overlaps with QK^T + softmax below.
  // D# fields per CDNA5 ISA 8.3/8.4 (2-D tile, groups 2/3 unused -> zero).
  {
    const unsigned lds_base = (unsigned)(unsigned long long)(void*)vband;
    const unsigned long long ga = (unsigned long long)(size_t)(vth + jlo);
    v4u g0;
    g0[0] = 1u;                                        // count=1, user mode
    g0[1] = lds_base;                                  // lds_addr
    g0[2] = (unsigned)(ga & 0xFFFFFFFFu);              // global_addr[31:0]
    g0[3] = (unsigned)(ga >> 32) | 0x80000000u;        // global_addr[56:32] | type=2
    v8i g1;
    g1[0] = 0x00010000;                                // data_size=2B, mask=0
    g1[1] = (int)(((unsigned)width & 0xFFFFu) << 16);  // tensor_dim0[15:0]
    g1[2] = (int)((((unsigned)width >> 16) & 0xFFFFu) | (64u << 16)); // dim0 hi|dim1 lo
    g1[3] = (int)(((unsigned)width & 0xFFFFu) << 16);  // dim1 hi=0 | tile_dim0
    g1[4] = 64;                                        // tile_dim1=64, tile_dim2=0
    g1[5] = SEQ;                                       // tensor_dim0_stride lo
    g1[6] = 0;                                         // stride0 hi | stride1 lo
    g1[7] = 0;                                         // stride1 hi
    v4i zz4 = {0, 0, 0, 0};
    v8i zz8 = {0, 0, 0, 0, 0, 0, 0, 0};
    __builtin_amdgcn_tensor_load_to_lds(g0, g1, zz4, zz4, zz8, 0);
  }
#endif

  const v16bf aq0 = frag_rowK(qh + (size_t)i0 * DH, DH, lane);
  const v16bf aq1 = frag_rowK(qh + (size_t)i0 * DH + 32, DH, lane);

  v8f st[NT];
  v8f mrow;
#pragma unroll
  for (int v = 0; v < 8; ++v) mrow[v] = -3.0e38f;

  // Pass A: scores + mask + lane-wise running max
#pragma unroll
  for (int jt = 0; jt < NT; ++jt) {
    const int tj = ti - (NT - 1) + jt;   // uniform per wave
    v8f s;
#pragma unroll
    for (int v = 0; v < 8; ++v) s[v] = -3.0e38f;
    if (tj >= 0) {
      const int j0 = tj * 16;
      const v16bf bk0 = frag_colN(kh + (size_t)j0 * DH, DH, lane);
      const v16bf bk1 = frag_colN(kh + (size_t)j0 * DH + 32, DH, lane);
      v8f acc = {};
      acc = wmma_bf16(aq0, bk0, acc);
      acc = wmma_bf16(aq1, bk1, acc);
      const int j = j0 + nn;
#pragma unroll
      for (int v = 0; v < 8; ++v) {
        const int i = i0 + v + 8 * half;
        const bool valid = (j <= i) && (j >= i - (WIN - 1));
        s[v] = valid ? acc[v] : -3.0e38f;
        mrow[v] = fmaxf(mrow[v], s[v]);
      }
    }
    st[jt] = s;
  }
  // cross-lane max within each 16-lane half (rows split lane0-15/16-31)
#pragma unroll
  for (int off = 1; off < 16; off <<= 1)
#pragma unroll
    for (int v = 0; v < 8; ++v)
      mrow[v] = fmaxf(mrow[v], __shfl_xor(mrow[v], off, 32));

  // Pass B: exp + row sums
  v8f lsum = {};
#pragma unroll
  for (int jt = 0; jt < NT; ++jt)
#pragma unroll
    for (int v = 0; v < 8; ++v) {
      const float p = __expf(st[jt][v] - mrow[v]);  // masked -> exp(-inf) = 0
      st[jt][v] = p;
      lsum[v] += p;
    }
#pragma unroll
  for (int off = 1; off < 16; off <<= 1)
#pragma unroll
    for (int v = 0; v < 8; ++v)
      lsum[v] += __shfl_xor(lsum[v], off, 32);
  v8f inv;
#pragma unroll
  for (int v = 0; v < 8; ++v) inv[v] = 1.0f / lsum[v];

  // Pass C: normalize, write fp32 attn band + bf16 LDS band
#pragma unroll
  for (int jt = 0; jt < NT; ++jt) {
    const int tj = ti - (NT - 1) + jt;
#pragma unroll
    for (int v = 0; v < 8; ++v) {
      const float p = st[jt][v] * inv[v];
      const int row = v + 8 * half;
      pband[row * LDP + jt * 16 + nn] = (__bf16)p;
      if (tj >= 0) ah[(size_t)(i0 + row) * SEQ + tj * 16 + nn] = p;
    }
  }
#if USE_TDM
  __builtin_amdgcn_s_wait_tensorcnt(0);
#endif
  __syncthreads();

  // O = P @ V : 9 K=32 pairs x 4 d-tiles; A from LDS P-band, B from V-band
  v8f o0 = {}, o1 = {}, o2 = {}, o3 = {};
#pragma unroll
  for (int pp = 0; pp < NT / 2; ++pp) {
    const int jbase = i0 - (NT - 1) * 16 + pp * 32;   // uniform
    if (jbase + 31 < 0) continue;                      // pair fully masked: P==0
    const v16bf ap = frag_rowK(&pband[pp * 32], LDP, lane);
#if USE_TDM
    int jcol = jbase - jlo + half * 16;
    if (jcol < 0) jcol = 0;                            // clamped rows hit zero P
#else
    int jl = jbase + half * 16;
    if (jl < 0) jl = 0;
#endif
#pragma unroll
    for (int dt = 0; dt < 4; ++dt) {
#if USE_TDM
      const __bf16* p = vband + (size_t)(dt * 16 + nn) * width + jcol;
#else
      const __bf16* p = vth + (size_t)(dt * 16 + nn) * SEQ + jl;
#endif
      v8bf lo = *(const v8bf*)p;
      v8bf hi = *(const v8bf*)(p + 8);
      v16bf bv;
#pragma unroll
      for (int e = 0; e < 8; ++e) { bv[e] = lo[e]; bv[8 + e] = hi[e]; }
      if (dt == 0) o0 = wmma_bf16(ap, bv, o0);
      else if (dt == 1) o1 = wmma_bf16(ap, bv, o1);
      else if (dt == 2) o2 = wmma_bf16(ap, bv, o2);
      else o3 = wmma_bf16(ap, bv, o3);
    }
  }
  // ctx: (B, S, H*Dh) bf16 — directly the A matrix of the out-projection GEMM
#pragma unroll
  for (int v = 0; v < 8; ++v) {
    const int row = v + 8 * half;
    const size_t base = ((size_t)b * SEQ + (i0 + row)) * INNER + h * DH;
    ctx[base + 0 + nn]  = (__bf16)o0[v];
    ctx[base + 16 + nn] = (__bf16)o1[v];
    ctx[base + 32 + nn] = (__bf16)o2[v];
    ctx[base + 48 + nn] = (__bf16)o3[v];
  }
}

// ---- Stage 3: output projection, 32x64 per wave ----------------------------
__global__ __launch_bounds__(32)
void gemm_out(const __bf16* __restrict__ ctx, const __bf16* __restrict__ wT,
              float* __restrict__ out) {
  const int lane = threadIdx.x;
  const int m0 = blockIdx.x * (16 * GM);
  const int n0 = blockIdx.y * (16 * GN);
  v8f c[GM][GN] = {};
  for (int k0 = 0; k0 < INNER; k0 += 32) {
    __builtin_prefetch(ctx + (size_t)m0 * INNER + k0 + 64, 0, 1);
    v16bf a[GM], b[GN];
#pragma unroll
    for (int mt = 0; mt < GM; ++mt)
      a[mt] = frag_rowK(ctx + (size_t)(m0 + 16 * mt) * INNER + k0, INNER, lane);
#pragma unroll
    for (int nt = 0; nt < GN; ++nt)
      b[nt] = frag_colN(wT + (size_t)(n0 + 16 * nt) * DMODEL + k0, DMODEL, lane);
#pragma unroll
    for (int mt = 0; mt < GM; ++mt)
#pragma unroll
      for (int nt = 0; nt < GN; ++nt)
        c[mt][nt] = wmma_bf16(a[mt], b[nt], c[mt][nt]);
  }
  const int half = lane >> 4;
#pragma unroll
  for (int nt = 0; nt < GN; ++nt) {
    const int col = n0 + nt * 16 + (lane & 15);
#pragma unroll
    for (int mt = 0; mt < GM; ++mt)
#pragma unroll
      for (int v = 0; v < 8; ++v) {
        const int row = m0 + mt * 16 + v + 8 * half;
        out[(size_t)row * DMODEL + col] = c[mt][nt][v];
      }
  }
}

// ---- Host launcher ---------------------------------------------------------
extern "C" void kernel_launch(void* const* d_in, const int* in_sizes, int n_in,
                              void* d_out, int out_size, void* d_ws, size_t ws_size,
                              hipStream_t stream) {
  const float* x    = (const float*)d_in[0];
  const float* Wqkv = (const float*)d_in[1];
  const float* Wout = (const float*)d_in[2];
  float* out  = (float*)d_out;
  float* attn = out + (size_t)NB * SEQ * DMODEL;

  char* ws = (char*)d_ws;
  size_t off = 0;
  auto carve = [&](size_t bytes) {
    void* p = ws + off;
    off += (bytes + 255) & ~(size_t)255;
    return p;
  };
  __bf16* xb  = (__bf16*)carve((size_t)BS * DMODEL * 2);
  __bf16* wqT = (__bf16*)carve((size_t)NQKV * DMODEL * 2);
  __bf16* woT = (__bf16*)carve((size_t)DMODEL * DMODEL * 2);
  __bf16* qb  = (__bf16*)carve((size_t)NB * HEADS * SEQ * DH * 2);
  __bf16* kb  = (__bf16*)carve((size_t)NB * HEADS * SEQ * DH * 2);
  __bf16* vtb = (__bf16*)carve((size_t)NB * HEADS * SEQ * DH * 2);
  __bf16* ctx = (__bf16*)carve((size_t)BS * INNER * 2);

  // 1) zero the (mostly sparse) attn output region
  zero_f4<<<4096, 256, 0, stream>>>((float4*)attn,
                                    ((size_t)NB * HEADS * SEQ * SEQ) / 4);
  // 2) precision conversion / weight transposes
  f32_to_bf16<<<2048, 256, 0, stream>>>(x, xb, BS * DMODEL);
  transpose_bf16<<<2048, 256, 0, stream>>>(Wqkv, wqT, DMODEL, NQKV);
  transpose_bf16<<<1024, 256, 0, stream>>>(Wout, woT, DMODEL, DMODEL);
  // 3) QKV projection
  gemm_qkv<<<dim3(BS / (16 * GM), NQKV / (16 * GN)), 32, 0, stream>>>(xb, wqT, qb, kb, vtb);
  // 4) banded attention (TDM-staged V, writes normalized attn band)
  attn_kernel<<<NB * HEADS * (SEQ / 16), 32, 0, stream>>>(qb, kb, vtb, attn, ctx);
  // 5) output projection
  gemm_out<<<dim3(BS / (16 * GM), DMODEL / (16 * GN)), 32, 0, stream>>>(ctx, woT, out);
}
